// PointNetFeaturePropagation_46420006535335
// MI455X (gfx1250) — matrix operand order repaired and checked by reading, not compiled, not run on hardware
//
#include <hip/hip_runtime.h>

// ---------------------------------------------------------------------------
// PointNet Feature Propagation for MI455X (gfx1250, wave32, WMMA).
// bf16 WMMA (16x16x32, f32 accum) for both MLP layers; KNN+interp fused into
// the GEMM0 LDS tile build so the 384-channel concat tensor never hits memory.
// ---------------------------------------------------------------------------

typedef __attribute__((ext_vector_type(16))) __bf16 v16bf;
typedef __attribute__((ext_vector_type(8)))  float  v8f;

#define BB   16
#define NN   8192
#define SS   2048
#define C1C  128
#define C2C  256
#define CIN  384     // C1 + C2
#define H0   256
#define H1   128
#define XLD  392     // padded LDS row stride (halves) for the 384-wide X tile
#define YLD  264     // padded LDS row stride (halves) for the 256-wide Y tile

union Frag { uint4 q[2]; v16bf v; };

__device__ __forceinline__ unsigned short f2bf(float f) {
  union { float f; unsigned u; } x; x.f = f;
  unsigned r = x.u + 0x7fffu + ((x.u >> 16) & 1u);   // round-to-nearest-even
  return (unsigned short)(r >> 16);
}

// ---------------------------------------------------------------------------
// Kernel 1: fold BN into per-channel scale/bias, convert weights to bf16.
// ---------------------------------------------------------------------------
__global__ void prep_kernel(const float* __restrict__ w0, const float* __restrict__ b0,
                            const float* __restrict__ g0, const float* __restrict__ be0,
                            const float* __restrict__ m0, const float* __restrict__ v0,
                            const float* __restrict__ w1, const float* __restrict__ b1,
                            const float* __restrict__ g1, const float* __restrict__ be1,
                            const float* __restrict__ m1, const float* __restrict__ v1,
                            unsigned short* __restrict__ w0bf, unsigned short* __restrict__ w1bf,
                            float* __restrict__ s0, float* __restrict__ t0,
                            float* __restrict__ s1, float* __restrict__ t1) {
  int i = blockIdx.x * 256 + threadIdx.x;          // grid covers H0*CIN = 98304
  if (i < H0 * CIN) w0bf[i] = f2bf(w0[i]);
  if (i < H1 * H0)  w1bf[i] = f2bf(w1[i]);
  if (i < H0) { float s = g0[i] * rsqrtf(v0[i] + 1e-5f); s0[i] = s; t0[i] = s * (b0[i] - m0[i]) + be0[i]; }
  if (i < H1) { float s = g1[i] * rsqrtf(v1[i] + 1e-5f); s1[i] = s; t1[i] = s * (b1[i] - m1[i]) + be1[i]; }
}

// ---------------------------------------------------------------------------
// Kernel 2: 3-NN search + inverse-distance weights.
// One block = 256 points of one batch; coords_2 of the batch staged in LDS.
// ---------------------------------------------------------------------------
__global__ void __launch_bounds__(256)
knn_kernel(const float* __restrict__ c1, const float* __restrict__ c2,
           int* __restrict__ idx, float* __restrict__ wgt) {
  __shared__ float sx[SS], sy[SS], sz[SS];         // 24 KB
  int b = blockIdx.x >> 5;
  int n = ((blockIdx.x & 31) << 8) + threadIdx.x;
  const float* cb = c2 + (long)b * 3 * SS;
  for (int i = threadIdx.x; i < SS; i += 256) {
    sx[i] = cb[i]; sy[i] = cb[SS + i]; sz[i] = cb[2 * SS + i];
  }
  __syncthreads();
  const float* pb = c1 + (long)b * 3 * NN;
  float px = pb[n], py = pb[NN + n], pz = pb[2 * NN + n];
  float d0 = 1e30f, d1 = 1e30f, d2 = 1e30f;
  int   i0 = 0, i1 = 0, i2 = 0;
  for (int s = 0; s < SS; ++s) {
    float dx = px - sx[s], dy = py - sy[s], dz = pz - sz[s];
    float d = dx * dx + dy * dy + dz * dz;
    if (d < d2) {
      if (d < d1) {
        if (d < d0) { d2 = d1; i2 = i1; d1 = d0; i1 = i0; d0 = d; i0 = s; }
        else        { d2 = d1; i2 = i1; d1 = d;  i1 = s; }
      } else        { d2 = d;  i2 = s; }
    }
  }
  d0 = fmaxf(d0, 1e-10f); d1 = fmaxf(d1, 1e-10f); d2 = fmaxf(d2, 1e-10f);
  float r0 = 1.f / d0, r1 = 1.f / d1, r2 = 1.f / d2;
  float rs = 1.f / (r0 + r1 + r2);
  long base = ((long)b * NN + n) * 3;
  idx[base + 0] = i0; idx[base + 1] = i1; idx[base + 2] = i2;
  wgt[base + 0] = r0 * rs; wgt[base + 1] = r1 * rs; wgt[base + 2] = r2 * rs;
}

// ---------------------------------------------------------------------------
// Kernel 3: layer 0 GEMM, interpolation fused into the LDS X^T tile build.
// Block = 64 points x all 256 output channels; 8 waves, each 2 M-tiles x 4 N-tiles.
// Y0 stored transposed [n][o] in bf16 == layer-1 B-fragment layout.
// ---------------------------------------------------------------------------
__global__ void __launch_bounds__(256)
gemm0_kernel(const float* __restrict__ f1, const float* __restrict__ f2,
             const int* __restrict__ idx, const float* __restrict__ wgt,
             const unsigned short* __restrict__ w0bf,
             const float* __restrict__ s0, const float* __restrict__ t0,
             unsigned short* __restrict__ y0) {
  __shared__ __attribute__((aligned(16))) unsigned short xt[64 * XLD];  // ~49 KB
  int b  = blockIdx.x >> 7;
  int n0 = (blockIdx.x & 127) << 6;
  int tid = threadIdx.x;

  // ---- phase 1: build X^T tile (64 points x 384 channels) in bf16 ----
  {
    int nl  = tid & 63;
    int cst = tid >> 6;                            // 0..3
    int n   = n0 + nl;
    long ib = ((long)b * NN + n) * 3;
    int  j0 = idx[ib], j1 = idx[ib + 1], j2 = idx[ib + 2];
    float wa = wgt[ib], wb = wgt[ib + 1], wc = wgt[ib + 2];
    const float* f1b = f1 + (long)b * C1C * NN;
    const float* f2b = f2 + (long)b * C2C * SS;
    for (int c = cst; c < CIN; c += 4) {
      float val;
      if (c < C1C) {
        val = f1b[(long)c * NN + n];
      } else {
        const float* r = f2b + (long)(c - C1C) * SS;
        val = wa * r[j0] + wb * r[j1] + wc * r[j2];
      }
      xt[nl * XLD + c] = f2bf(val);
    }
  }
  __syncthreads();

  // ---- phase 2: WMMA GEMM  D(256x64) = W0(256x384) * X(384x64) ----
  int wv = tid >> 5, lane = tid & 31;
  int lm = lane & 15, half = lane >> 4;
  int mA = (wv * 2) * 16, mB = mA + 16;
  const unsigned short* a0p = w0bf + (long)(mA + lm) * CIN + half * 8;
  const unsigned short* a1p = w0bf + (long)(mB + lm) * CIN + half * 8;

  v8f zero = {0.f, 0.f, 0.f, 0.f, 0.f, 0.f, 0.f, 0.f};
  v8f acc0[4], acc1[4];
  for (int j = 0; j < 4; ++j) { acc0[j] = zero; acc1[j] = zero; }

  for (int ks = 0; ks < 12; ++ks) {
    int kb = ks * 32;
    Frag a0, a1;
    a0.q[0] = *(const uint4*)(a0p + kb);
    a0.q[1] = *(const uint4*)(a0p + kb + 16);
    a1.q[0] = *(const uint4*)(a1p + kb);
    a1.q[1] = *(const uint4*)(a1p + kb + 16);
    for (int j = 0; j < 4; ++j) {
      const unsigned short* bp = xt + (16 * j + lm) * XLD + kb + half * 8;
      Frag bb;
      bb.q[0] = *(const uint4*)(bp);
      bb.q[1] = *(const uint4*)(bp + 16);
      acc0[j] = __builtin_amdgcn_wmma_f32_16x16x32_bf16(false, a0.v, false, bb.v, (short)0, acc0[j], false, false);
      acc1[j] = __builtin_amdgcn_wmma_f32_16x16x32_bf16(false, a1.v, false, bb.v, (short)0, acc1[j], false, false);
    }
  }

  // ---- epilogue: BN + ReLU, store Y0^T[n][o] as bf16 (one b128 per lane/tile)
  for (int j = 0; j < 4; ++j) {
    int n = n0 + 16 * j + lm;
    unsigned short* yrow = y0 + ((long)b * NN + n) * H0;
    {
      int ob = mA + half * 8;
      union { uint4 q; unsigned short s[8]; } pk;
      for (int r = 0; r < 8; ++r)
        pk.s[r] = f2bf(fmaxf(s0[ob + r] * acc0[j][r] + t0[ob + r], 0.f));
      *(uint4*)(yrow + ob) = pk.q;
    }
    {
      int ob = mB + half * 8;
      union { uint4 q; unsigned short s[8]; } pk;
      for (int r = 0; r < 8; ++r)
        pk.s[r] = f2bf(fmaxf(s0[ob + r] * acc1[j][r] + t0[ob + r], 0.f));
      *(uint4*)(yrow + ob) = pk.q;
    }
  }
}

// ---------------------------------------------------------------------------
// Kernel 4: layer 1 GEMM  out(128xN) = W1(128x256) * Y0(256xN), BN+ReLU, f32 out.
// ---------------------------------------------------------------------------
__global__ void __launch_bounds__(256)
gemm1_kernel(const unsigned short* __restrict__ y0,
             const unsigned short* __restrict__ w1bf,
             const float* __restrict__ s1, const float* __restrict__ t1,
             float* __restrict__ out) {
  __shared__ __attribute__((aligned(16))) unsigned short yt[64 * YLD];  // ~33 KB
  int b  = blockIdx.x >> 7;
  int n0 = (blockIdx.x & 127) << 6;
  int tid = threadIdx.x;

  // stage Y0^T tile (64 points x 256 ch) into padded LDS
  for (int i = tid; i < 64 * 32; i += 256) {
    int nl = i >> 5, ch = i & 31;
    const uint4* src = (const uint4*)(y0 + ((long)b * NN + n0 + nl) * H0);
    *(uint4*)(yt + nl * YLD + ch * 8) = src[ch];
  }
  __syncthreads();

  int wv = tid >> 5, lane = tid & 31;
  int lm = lane & 15, half = lane >> 4;
  int mrow = wv * 16;                               // 8 waves cover M=128
  const unsigned short* ap = w1bf + (long)(mrow + lm) * H0 + half * 8;

  v8f zero = {0.f, 0.f, 0.f, 0.f, 0.f, 0.f, 0.f, 0.f};
  v8f acc[4];
  for (int j = 0; j < 4; ++j) acc[j] = zero;

  for (int ks = 0; ks < 8; ++ks) {
    int kb = ks * 32;
    Frag a;
    a.q[0] = *(const uint4*)(ap + kb);
    a.q[1] = *(const uint4*)(ap + kb + 16);
    for (int j = 0; j < 4; ++j) {
      const unsigned short* bp = yt + (16 * j + lm) * YLD + kb + half * 8;
      Frag bb;
      bb.q[0] = *(const uint4*)(bp);
      bb.q[1] = *(const uint4*)(bp + 16);
      acc[j] = __builtin_amdgcn_wmma_f32_16x16x32_bf16(false, a.v, false, bb.v, (short)0, acc[j], false, false);
    }
  }

  for (int j = 0; j < 4; ++j) {
    int n  = n0 + 16 * j + lm;
    int ob = mrow + half * 8;
    for (int r = 0; r < 8; ++r) {
      float y = s1[ob + r] * acc[j][r] + t1[ob + r];
      out[((long)b * H1 + ob + r) * NN + n] = fmaxf(y, 0.f);
    }
  }
}

// ---------------------------------------------------------------------------
// Host launcher.  Workspace layout (bytes, 16B-aligned):
//   0        : W0 bf16          (196608)
//   196608   : W1 bf16          ( 65536)
//   262144   : s0 (1024), 263168: t0 (1024), 264192: s1 (512), 264704: t1 (512)
//   265216   : idx  int32 B*N*3 (1572864)
//   1838080  : wgt  f32   B*N*3 (1572864)
//   3410944  : Y0^T bf16 B*N*256 (67108864)   => total ~70.5 MB
// ---------------------------------------------------------------------------
extern "C" void kernel_launch(void* const* d_in, const int* in_sizes, int n_in,
                              void* d_out, int out_size, void* d_ws, size_t ws_size,
                              hipStream_t stream) {
  const float* c1  = (const float*)d_in[0];
  const float* c2  = (const float*)d_in[1];
  const float* f1  = (const float*)d_in[2];
  const float* f2  = (const float*)d_in[3];
  const float* w0  = (const float*)d_in[4];
  const float* b0  = (const float*)d_in[5];
  const float* g0  = (const float*)d_in[6];
  const float* be0 = (const float*)d_in[7];
  const float* m0  = (const float*)d_in[8];
  const float* v0  = (const float*)d_in[9];
  const float* w1  = (const float*)d_in[10];
  const float* b1  = (const float*)d_in[11];
  const float* g1  = (const float*)d_in[12];
  const float* be1 = (const float*)d_in[13];
  const float* m1  = (const float*)d_in[14];
  const float* v1  = (const float*)d_in[15];

  char* ws = (char*)d_ws;
  unsigned short* w0bf = (unsigned short*)(ws + 0);
  unsigned short* w1bf = (unsigned short*)(ws + 196608);
  float* s0 = (float*)(ws + 262144);
  float* t0 = (float*)(ws + 263168);
  float* s1 = (float*)(ws + 264192);
  float* t1 = (float*)(ws + 264704);
  int*   idx = (int*)(ws + 265216);
  float* wgt = (float*)(ws + 1838080);
  unsigned short* y0 = (unsigned short*)(ws + 3410944);

  prep_kernel<<<384, 256, 0, stream>>>(w0, b0, g0, be0, m0, v0,
                                       w1, b1, g1, be1, m1, v1,
                                       w0bf, w1bf, s0, t0, s1, t1);
  knn_kernel<<<512, 256, 0, stream>>>(c1, c2, idx, wgt);
  gemm0_kernel<<<2048, 256, 0, stream>>>(f1, f2, idx, wgt, w0bf, s0, t0, y0);
  gemm1_kernel<<<2048, 256, 0, stream>>>(y0, w1bf, s1, t1, (float*)d_out);
}
